// KPDecoder_2370821948161
// MI455X (gfx1250) — matrix operand order repaired and checked by reading, not compiled, not run on hardware
//
#include <hip/hip_runtime.h>
#include <hip/hip_bf16.h>

// ---------------------------------------------------------------------------
// MI455X (gfx1250) KPConv decoder.
// bf16 WMMA (v_wmma_f32_16x16x32_bf16) with f32 accumulate; gather+concat into
// LDS (bank-conflict-free padded layout); weights pre-transposed to bf16 [N][K]
// so B fragments are two 16B global loads per lane, double-buffered so WMMA never
// waits on an in-flight load. GroupNorm+LeakyReLU fused in-workgroup.
// ---------------------------------------------------------------------------

typedef __attribute__((ext_vector_type(16))) __bf16 v16bf;
typedef __attribute__((ext_vector_type(8)))  float  v8f;
typedef __attribute__((ext_vector_type(4)))  float  f32x4;

struct Frag32 { uint4 lo; uint4 hi; };

static __device__ __forceinline__ unsigned short f2bf(float f) {
    unsigned int u = __builtin_bit_cast(unsigned int, f);
    u += 0x7FFFu + ((u >> 16) & 1u);           // round-to-nearest-even
    return (unsigned short)(u >> 16);
}

static __device__ __forceinline__ v16bf make_frag(const unsigned short* p0,
                                                  const unsigned short* p1) {
    Frag32 s;
    s.lo = *(const uint4*)p0;
    s.hi = *(const uint4*)p1;
    return __builtin_bit_cast(v16bf, s);
}

// --- weight convert + transpose: W [K][N] f32  ->  Wt [N][K] bf16 -----------
__global__ void wt_convert(const float* __restrict__ W,
                           unsigned short* __restrict__ Wt, int K, int N) {
    int i = blockIdx.x * blockDim.x + threadIdx.x;   // i = n*K + k
    if (i >= N * K) return;
    int n = i / K;
    int k = i - n * K;
    Wt[i] = f2bf(W[(size_t)k * N + n]);
}

// --- fused decoder stage -----------------------------------------------------
// Y[M x N] = GN/LReLU( concat(srcA[up[m,0]], srcB[m]) @ W + b )
template <int K_, int KA_, int N_, int GS_, bool GN_>
__global__ __launch_bounds__(256)
void decoder_stage(const float* __restrict__ srcA,   // coarse feats [*, KA_]
                   const float* __restrict__ srcB,   // fine feats   [M, KB_]
                   const int*   __restrict__ up,     // [M, 40] neighbor idx
                   const unsigned short* __restrict__ Wt, // bf16 [N_][K_]
                   const float* __restrict__ bias,
                   const float* __restrict__ gamma,
                   const float* __restrict__ beta,
                   float* __restrict__ out, int M) {
    constexpr int KB_  = K_ - KA_;
    constexpr int LDSK = K_ + 8;     // bf16 elems per LDS row (pad kills bank conflicts)
    constexpr int NY   = N_ + 4;     // f32 elems per output LDS row
    constexpr int TNW  = N_ / 128;   // 16-col tiles per wave (8 waves)
    constexpr int KT   = K_ / 32;    // k-steps (24 or 12: always even)

    extern __shared__ char smem[];
    unsigned short* smX   = (unsigned short*)smem;
    float*          smY   = (float*)(smem + (size_t)32 * LDSK * 2);
    int*            smIdx = (int*)(smem + (size_t)32 * LDSK * 2 + (size_t)32 * NY * 4);

    const int tid  = threadIdx.x;
    const int wave = tid >> 5;
    const int lane = tid & 31;
    const int lan  = lane & 15;
    const int hh   = lane >> 4;
    const int m0   = blockIdx.x * 32;

    // stage the gather indices (nearest neighbor = column 0 of 40)
    if (tid < 32) {
        int r = m0 + tid;
        if (r >= M) r = M - 1;
        smIdx[tid] = up[(size_t)r * 40];
    }
    __syncthreads();

    // --- load X tile: gather + concat + f32->bf16 into LDS ------------------
    constexpr int CH = K_ / 4;                 // float4 chunks per row
    for (int c = tid; c < 32 * CH; c += 256) {
        int row = c / CH;
        int e   = (c - row * CH) * 4;
        int gr  = m0 + row; if (gr >= M) gr = M - 1;
        const float* src = (e < KA_)
            ? srcA + (size_t)smIdx[row] * KA_ + e
            : srcB + (size_t)gr * KB_ + (e - KA_);
        float4 v = *(const float4*)src;
        uint2 pk;
        pk.x = (unsigned)f2bf(v.x) | ((unsigned)f2bf(v.y) << 16);
        pk.y = (unsigned)f2bf(v.z) | ((unsigned)f2bf(v.w) << 16);
        *(uint2*)&smX[(size_t)row * LDSK + e] = pk;
    }
    __syncthreads();

    // --- GEMM: 2 M-tiles x TNW N-tiles per wave, K steps of 32,
    //     B fragments double-buffered (ping/pong) --------------------------
    const int nbase = wave * (N_ / 8);
    v8f acc[2][TNW] = {};

    auto loadB = [&](v16bf* b, int k0) {
#pragma unroll
        for (int nt = 0; nt < TNW; ++nt) {
            const unsigned short* q =
                Wt + (size_t)(nbase + nt * 16 + lan) * K_ + k0 + 8 * hh;
            b[nt] = make_frag(q, q + 16);      // two 16B global loads (k 8h..8h+7, 16+8h..)
        }
    };
    auto loadA = [&](v16bf* a, int k0) {
#pragma unroll
        for (int mt = 0; mt < 2; ++mt) {
            const unsigned short* p =
                &smX[(size_t)(mt * 16 + lan) * LDSK + k0 + 8 * hh];
            a[mt] = make_frag(p, p + 16);      // ds_load_b128 x2, conflict-free
        }
    };
    auto mm = [&](const v16bf* a, const v16bf* b) {
#pragma unroll
        for (int nt = 0; nt < TNW; ++nt) {
            acc[0][nt] = __builtin_amdgcn_wmma_f32_16x16x32_bf16(
                false, a[0], false, b[nt], (short)0, acc[0][nt], false, false);
            acc[1][nt] = __builtin_amdgcn_wmma_f32_16x16x32_bf16(
                false, a[1], false, b[nt], (short)0, acc[1][nt], false, false);
        }
    };

    v16bf b0[TNW], b1[TNW], a[2];
    loadB(b0, 0);                              // prologue fill
    for (int kt = 0; kt < KT; kt += 2) {
        const int k0 = kt * 32;
        loadA(a, k0);
        loadB(b1, k0 + 32);                    // next step's B in flight
        __builtin_prefetch(&Wt[(size_t)(nbase + lan) * K_ + k0 + 64], 0, 1);
        mm(a, b0);                             // waits only on last iter's loads
        loadA(a, k0 + 32);
        if (kt + 2 < KT) loadB(b0, k0 + 64);
        mm(a, b1);
    }

    // --- epilogue: bias into LDS f32 tile ------------------------------------
#pragma unroll
    for (int mt = 0; mt < 2; ++mt) {
#pragma unroll
        for (int nt = 0; nt < TNW; ++nt) {
            const int col = nbase + nt * 16 + lan;
            const float bv = bias[col];
#pragma unroll
            for (int r = 0; r < 8; ++r) {
                const int row = mt * 16 + hh * 8 + r;   // D-tile VGPR->row map
                smY[(size_t)row * NY + col] = acc[mt][nt][r] + bv;
            }
        }
    }
    __syncthreads();

    // --- GroupNorm(32) + LeakyReLU(0.1), per (row, group) --------------------
    if (GN_) {
        for (int t = tid; t < 32 * 32; t += 256) {
            const int row = t >> 5;
            const int g   = t & 31;
            float* p = &smY[(size_t)row * NY + g * GS_];
            float s = 0.f, s2 = 0.f;
#pragma unroll
            for (int i = 0; i < GS_; ++i) { float x = p[i]; s += x; s2 += x * x; }
            const float mean = s * (1.f / GS_);
            const float var  = s2 * (1.f / GS_) - mean * mean;
            const float inv  = rsqrtf(var + 1e-5f);
#pragma unroll
            for (int i = 0; i < GS_; ++i) {
                const int c = g * GS_ + i;
                float y = (p[i] - mean) * inv * gamma[c] + beta[c];
                p[i] = y >= 0.f ? y : 0.1f * y;
            }
        }
        __syncthreads();
    }

    // --- vectorized writeback -------------------------------------------------
    // GN stages feed the next stage's gather -> keep RT (L2-resident).
    // Final stage output is write-once -> non-temporal, don't pollute L2.
    for (int c = tid; c < 32 * (N_ / 4); c += 256) {
        const int row = c / (N_ / 4);
        const int q4  = (c - row * (N_ / 4)) * 4;
        const int gr  = m0 + row;
        if (gr < M) {
            f32x4 v = *(const f32x4*)&smY[(size_t)row * NY + q4];
            f32x4* dst = (f32x4*)&out[(size_t)gr * N_ + q4];
            if (GN_) *dst = v;
            else     __builtin_nontemporal_store(v, dst);
        }
    }
}

// ---------------------------------------------------------------------------
extern "C" void kernel_launch(void* const* d_in, const int* in_sizes, int n_in,
                              void* d_out, int out_size, void* d_ws, size_t ws_size,
                              hipStream_t stream) {
    const float* f1  = (const float*)d_in[0];
    const float* f2  = (const float*)d_in[1];
    const float* f3  = (const float*)d_in[2];
    const float* f4  = (const float*)d_in[3];
    const int*   up0 = (const int*)d_in[4];
    const int*   up1 = (const int*)d_in[5];
    const int*   up2 = (const int*)d_in[6];
    const float* W3  = (const float*)d_in[7];
    const float* b3  = (const float*)d_in[8];
    const float* g3  = (const float*)d_in[9];
    const float* be3 = (const float*)d_in[10];
    const float* W2  = (const float*)d_in[11];
    const float* b2  = (const float*)d_in[12];
    const float* g2  = (const float*)d_in[13];
    const float* be2 = (const float*)d_in[14];
    const float* W1  = (const float*)d_in[15];
    const float* b1  = (const float*)d_in[16];

    const int M1 = in_sizes[0] / 128;   // 200000
    const int M2 = in_sizes[1] / 256;   // 50000
    const int M3 = in_sizes[2] / 256;   // 12500

    float* o1 = (float*)d_out;
    float* o2 = o1 + (size_t)M1 * 128;
    float* o3 = o2 + (size_t)M2 * 256;

    unsigned short* wt3 = (unsigned short*)d_ws;
    unsigned short* wt2 = wt3 + (size_t)768 * 512;
    unsigned short* wt1 = wt2 + (size_t)768 * 256;

    // bf16 transposed weights (L2-resident; W3 is only 0.75 MB)
    wt_convert<<<(768 * 512 + 255) / 256, 256, 0, stream>>>(W3, wt3, 768, 512);
    wt_convert<<<(768 * 256 + 255) / 256, 256, 0, stream>>>(W2, wt2, 768, 256);
    wt_convert<<<(384 * 128 + 255) / 256, 256, 0, stream>>>(W1, wt1, 384, 128);

    const size_t shm3 = (size_t)32 * (768 + 8) * 2 + (size_t)32 * (512 + 4) * 4 + 128;
    const size_t shm2 = (size_t)32 * (768 + 8) * 2 + (size_t)32 * (256 + 4) * 4 + 128;
    const size_t shm1 = (size_t)32 * (384 + 8) * 2 + (size_t)32 * (128 + 4) * 4 + 128;

    (void)hipFuncSetAttribute((const void*)decoder_stage<768, 512, 512, 16, true>,
                              hipFuncAttributeMaxDynamicSharedMemorySize, (int)shm3);
    (void)hipFuncSetAttribute((const void*)decoder_stage<768, 512, 256, 8, true>,
                              hipFuncAttributeMaxDynamicSharedMemorySize, (int)shm2);
    (void)hipFuncSetAttribute((const void*)decoder_stage<384, 256, 128, 1, false>,
                              hipFuncAttributeMaxDynamicSharedMemorySize, (int)shm1);

    // stage 3: [N3, 768] @ [768, 512] + GN + LReLU
    decoder_stage<768, 512, 512, 16, true>
        <<<(M3 + 31) / 32, 256, shm3, stream>>>(f4, f3, up2, wt3, b3, g3, be3, o3, M3);
    // stage 2: [N2, 768] @ [768, 256] + GN + LReLU (consumes latent_s3)
    decoder_stage<768, 512, 256, 8, true>
        <<<(M2 + 31) / 32, 256, shm2, stream>>>(o3, f2, up1, wt2, b2, g2, be2, o2, M2);
    // stage 1: [N1, 384] @ [384, 128], linear only (consumes latent_s2)
    decoder_stage<384, 256, 128, 1, false>
        <<<(M1 + 31) / 32, 256, shm1, stream>>>(o2, f1, up0, wt1, b1, nullptr, nullptr, o1, M1);
}
